// Linear_int8_21663815041266
// MI455X (gfx1250) — 8-bit matrix operand order repaired and checked by reading, compiled, not run on hardware
//
#include <hip/hip_runtime.h>

typedef int v8i __attribute__((ext_vector_type(8)));

#define NDIM 768
#define MDIM 256

__global__ void init_ws_kernel(unsigned int* maxbits) {
    maxbits[0] = 0u;
    maxbits[1] = 0u;
}

__global__ __launch_bounds__(256) void absmax_kernel(const float* __restrict__ p, int n,
                                                     unsigned int* __restrict__ target) {
    __shared__ unsigned int red[256];
    unsigned int m = 0u;
    for (int i = blockIdx.x * blockDim.x + threadIdx.x; i < n; i += gridDim.x * blockDim.x) {
        unsigned int b = __float_as_uint(p[i]) & 0x7fffffffu;
        m = (b > m) ? b : m;
    }
    red[threadIdx.x] = m;
    __syncthreads();
    for (int s = 128; s > 0; s >>= 1) {
        if ((int)threadIdx.x < s) {
            unsigned int o = red[threadIdx.x + s];
            if (o > red[threadIdx.x]) red[threadIdx.x] = o;
        }
        __syncthreads();
    }
    if (threadIdx.x == 0) atomicMax(target, red[0]);
}

__global__ __launch_bounds__(256) void quantize_kernel(const float* __restrict__ p, int n,
                                                       const unsigned int* __restrict__ maxbits,
                                                       signed char* __restrict__ q) {
    float scale = __uint_as_float(*maxbits) / 127.0f;
    int i = blockIdx.x * blockDim.x + threadIdx.x;
    if (i < n) {
        float v = rintf(p[i] / scale);
        v = fminf(127.0f, fmaxf(-127.0f, v));
        q[i] = (signed char)(int)v;
    }
}

__global__ __launch_bounds__(256) void noise_kernel(const int* __restrict__ lut,
                                                    signed char* __restrict__ n8) {
    int i = blockIdx.x * blockDim.x + threadIdx.x;
    if (i < 65536) {
        int a = (i >> 8) - 128;
        int b = (i & 255) - 128;
        n8[i] = (signed char)(lut[i] - a * b);
    }
}

__global__ __launch_bounds__(256) void gemm_lut_kernel(
    const signed char* __restrict__ xq, const signed char* __restrict__ wq,
    const signed char* __restrict__ noise8, const unsigned int* __restrict__ maxbits,
    const float* __restrict__ bias, float* __restrict__ out) {

    __shared__ signed char lds_noise[65536];
    {
        const int4* src = (const int4*)noise8;
        int4* dst = (int4*)lds_noise;
        for (int i = threadIdx.x; i < 4096; i += 256) dst[i] = src[i];
    }
    __syncthreads();

    const int wave = threadIdx.x >> 5;
    const int lane = threadIdx.x & 31;
    const int l15  = lane & 15;
    const int hi   = lane >> 4;

    const int m0 = blockIdx.x * 32 + (wave >> 2) * 16;
    const int n0 = blockIdx.y * 64 + (wave & 3) * 16;

    const int mA = m0 + l15;
    const int nB = n0 + l15;
    const int mG = m0 + hi * 8;

    v8i c = {};
    int nacc[8];
#pragma unroll
    for (int r = 0; r < 8; r++) nacc[r] = 0;

    for (int k0 = 0; k0 < NDIM; k0 += 64) {
        const signed char* arow = xq + mA * NDIM + k0;
        const signed char* brow = wq + nB * NDIM + k0;
        if (k0 + 64 < NDIM) {
            __builtin_prefetch(arow + 64, 0, 0);
            __builtin_prefetch(brow + 64, 0, 0);
        }

        v8i a, b;
#pragma unroll
        for (int i = 0; i < 8; i++) {
            int koff = ((i >> 1) << 4) + ((i & 1) << 2) + (hi << 3);
            a[i] = *(const int*)(arow + koff);
        }
#pragma unroll
        for (int j = 0; j < 8; j++) {
            int koff = ((j >> 1) << 4) + ((j & 1) << 2) + (hi << 3);
            b[j] = *(const int*)(brow + koff);
        }
        c = __builtin_amdgcn_wmma_i32_16x16x64_iu8(true, a, true, b, c, false, false);

        const signed char* xbase = xq + mG * NDIM + k0;
        for (int kk = 0; kk < 64; kk += 4) {
            unsigned int wi4 = (*(const unsigned int*)(brow + kk)) ^ 0x80808080u;
            unsigned int xr[8];
#pragma unroll
            for (int r = 0; r < 8; r++)
                xr[r] = (*(const unsigned int*)(xbase + r * NDIM + kk)) ^ 0x80808080u;
#pragma unroll
            for (int t = 0; t < 4; t++) {
                unsigned int wb = (wi4 >> (t * 8)) & 0xFFu;
#pragma unroll
                for (int r = 0; r < 8; r++) {
                    unsigned int xb = (xr[r] >> (t * 8)) & 0xFFu;
                    nacc[r] += (int)lds_noise[(xb << 8) | wb];
                }
            }
        }
    }

    float sx = __uint_as_float(maxbits[0]) / 127.0f;
    float sw = __uint_as_float(maxbits[1]) / 127.0f;
    float s = sx * sw;

#pragma unroll
    for (int r = 0; r < 8; r++) {
        int M = mG + r;
        int N = nB;
        out[M * NDIM + N] = (float)(c[r] + nacc[r]) * s + bias[N];
    }
}

extern "C" void kernel_launch(void* const* d_in, const int* in_sizes, int n_in,
                              void* d_out, int out_size, void* d_ws, size_t ws_size,
                              hipStream_t stream) {
    const float* x    = (const float*)d_in[0];
    const float* w    = (const float*)d_in[1];
    const float* bias = (const float*)d_in[2];
    const int*   lut  = (const int*)d_in[3];
    float* out = (float*)d_out;

    char* ws = (char*)d_ws;
    unsigned int* maxbits = (unsigned int*)ws;
    signed char* xq = (signed char*)(ws + 64);
    signed char* wq = (signed char*)(ws + 64 + 196608);
    signed char* n8 = (signed char*)(ws + 64 + 196608 + 589824);

    const int NX = MDIM * NDIM;
    const int NW = NDIM * NDIM;

    init_ws_kernel<<<1, 1, 0, stream>>>(maxbits);
    absmax_kernel<<<96, 256, 0, stream>>>(x, NX, maxbits + 0);
    absmax_kernel<<<96, 256, 0, stream>>>(w, NW, maxbits + 1);
    quantize_kernel<<<(NX + 255) / 256, 256, 0, stream>>>(x, NX, maxbits + 0, xq);
    quantize_kernel<<<(NW + 255) / 256, 256, 0, stream>>>(w, NW, maxbits + 1, wq);
    noise_kernel<<<256, 256, 0, stream>>>(lut, n8);

    dim3 grid(MDIM / 32, NDIM / 64);
    gemm_lut_kernel<<<grid, 256, 0, stream>>>(xq, wq, n8, maxbits, bias, out);
}
